// SA_Attention_41532333752477
// MI455X (gfx1250) — matrix operand order repaired and checked by reading, not compiled, not run on hardware
//
#include <hip/hip_runtime.h>
#include <hip/hip_bf16.h>

// ---------------- problem constants (match reference) ----------------
#define B_      16
#define N_      2048
#define S_      512      // N_CENTER
#define K_      32       // N_NEAR
#define KNN_    10       // N_STEPK
#define D_      64       // D_FEA
#define C0PAD_  96       // 3+64 = 67 padded to 3*32
#define C1_     64
#define C2_     64
#define M2_     128
#define NTOT_   (B_ * S_ * K_)    // 262144 GEMM columns
#define BN_EPS_ 1e-5f

typedef __attribute__((ext_vector_type(16))) __bf16 v16bf;
typedef __attribute__((ext_vector_type(8)))  __bf16 v8bf;
typedef __attribute__((ext_vector_type(8)))  float  v8f;

// ---------------------------------------------------------------------
// 1) Brute-force 10-NN (+ per-row farthest index) from LDS-resident xyz
// ---------------------------------------------------------------------
__global__ __launch_bounds__(256) void knn_kernel(
    const float* __restrict__ xyz,   // [B,3,N]
    int* __restrict__ nbr,           // [B,N,10]
    int* __restrict__ maxind)        // [B,N]
{
  __shared__ float sx[N_], sy[N_], sz[N_];   // 24 KB of the 320 KB WGP LDS
  int b   = blockIdx.x >> 3;   // 8 blocks per batch
  int blk = blockIdx.x & 7;
  const float* xb = xyz + (size_t)b * 3 * N_;
  for (int i = threadIdx.x; i < N_; i += 256) {
    sx[i] = xb[i]; sy[i] = xb[N_ + i]; sz[i] = xb[2 * N_ + i];
  }
  __syncthreads();

  int n = blk * 256 + threadIdx.x;
  float cx = sx[n], cy = sy[n], cz = sz[n];
  float nd[KNN_ + 1]; int ni[KNN_ + 1];
  #pragma unroll
  for (int j = 0; j <= KNN_; ++j) { nd[j] = 3.0e38f; ni[j] = n; }
  float dmax = -1.f; int imax = 0;

  for (int m = 0; m < N_; ++m) {
    float dx = sx[m] - cx, dy = sy[m] - cy, dz = sz[m] - cz;
    float d = dx * dx + dy * dy + dz * dz;
    if (d > dmax) { dmax = d; imax = m; }          // first-occurrence argmax
    if (d < nd[KNN_]) {                            // stable k+1 smallest
      int j = KNN_;
      while (j > 0 && d < nd[j - 1]) { nd[j] = nd[j - 1]; ni[j] = ni[j - 1]; --j; }
      nd[j] = d; ni[j] = m;
    }
  }
  int* o = nbr + ((size_t)b * N_ + n) * KNN_;
  #pragma unroll
  for (int j = 0; j < KNN_; ++j) o[j] = ni[j + 1];  // drop self (slot 0)
  maxind[(size_t)b * N_ + n] = imax;
}

// ---------------------------------------------------------------------
// 2) Farthest point sampling: one block per batch, sequential 512 steps
//    Also writes new_xyz ([B,3,S]) directly into d_out.
// ---------------------------------------------------------------------
__global__ __launch_bounds__(256) void fps_kernel(
    const float* __restrict__ xyz, int* __restrict__ fpsidx, float* __restrict__ out)
{
  int b = blockIdx.x, tid = threadIdx.x;
  const float* xb = xyz + (size_t)b * 3 * N_;
  float px[8], py[8], pz[8], dv[8];
  #pragma unroll
  for (int j = 0; j < 8; ++j) {
    int p = tid + j * 256;
    px[j] = xb[p]; py[j] = xb[N_ + p]; pz[j] = xb[2 * N_ + p];
    dv[j] = 1e10f;
  }
  __shared__ float rv[256]; __shared__ int ri[256];
  __shared__ float cc[3];  __shared__ int curs;
  if (tid == 0) curs = 0;
  __syncthreads();

  for (int i = 0; i < S_; ++i) {
    int far = curs;
    if (tid == 0) {
      fpsidx[b * S_ + i] = far;
      cc[0] = xb[far]; cc[1] = xb[N_ + far]; cc[2] = xb[2 * N_ + far];
      out[(b * 3 + 0) * S_ + i] = cc[0];
      out[(b * 3 + 1) * S_ + i] = cc[1];
      out[(b * 3 + 2) * S_ + i] = cc[2];
    }
    __syncthreads();
    float bx = cc[0], by = cc[1], bz = cc[2];
    float bestv = -1.f; int besti = 0;
    #pragma unroll
    for (int j = 0; j < 8; ++j) {
      float dx = px[j] - bx, dy = py[j] - by, dz = pz[j] - bz;
      float d = dx * dx + dy * dy + dz * dz;
      dv[j] = fminf(dv[j], d);
      int p = tid + j * 256;
      if (dv[j] > bestv || (dv[j] == bestv && p < besti)) { bestv = dv[j]; besti = p; }
    }
    rv[tid] = bestv; ri[tid] = besti;
    __syncthreads();
    for (int off = 128; off > 0; off >>= 1) {
      if (tid < off) {
        float v2 = rv[tid + off]; int i2 = ri[tid + off];
        if (v2 > rv[tid] || (v2 == rv[tid] && i2 < ri[tid])) { rv[tid] = v2; ri[tid] = i2; }
      }
      __syncthreads();
    }
    if (tid == 0) curs = ri[0];
    __syncthreads();
  }
}

// ---------------------------------------------------------------------
// 3) surface-KNN expansion (one pass: 10x10 candidates) for the 512
//    selected centers only: sort, dedupe->max_ind, stable 33-smallest.
// ---------------------------------------------------------------------
__global__ __launch_bounds__(256) void expand_kernel(
    const float* __restrict__ xyz, const int* __restrict__ nbr,
    const int* __restrict__ maxind, const int* __restrict__ fpsidx,
    int* __restrict__ idx)           // [B,S,32]
{
  int t = blockIdx.x * 256 + threadIdx.x;
  if (t >= B_ * S_) return;
  int b = t / S_, s = t % S_;
  int n = fpsidx[b * S_ + s];

  int cand[KNN_ * KNN_];
  const int* nb = nbr + ((size_t)b * N_ + n) * KNN_;
  for (int j = 0; j < KNN_; ++j) {
    int c = nb[j];
    const int* nc = nbr + ((size_t)b * N_ + c) * KNN_;
    for (int u = 0; u < KNN_; ++u) cand[j * KNN_ + u] = nc[u];
  }
  // sort candidate indices
  for (int i = 1; i < KNN_ * KNN_; ++i) {
    int v = cand[i], j = i - 1;
    while (j >= 0 && cand[j] > v) { cand[j + 1] = cand[j]; --j; }
    cand[j + 1] = v;
  }
  // dedupe: duplicates replaced by the row's farthest point index
  int mi = maxind[(size_t)b * N_ + n];
  int prev = cand[0];
  for (int i = 1; i < KNN_ * KNN_; ++i) {
    int cur = cand[i];
    if (cur == prev) cand[i] = mi;
    prev = cur;
  }
  // stable selection of the 33 smallest distances (slot 0 == self, dropped)
  const float* xb = xyz + (size_t)b * 3 * N_;
  float cx = xb[n], cy = xb[N_ + n], cz = xb[2 * N_ + n];
  float bd[K_ + 1]; int bi[K_ + 1];
  #pragma unroll
  for (int i = 0; i <= K_; ++i) { bd[i] = 3.0e38f; bi[i] = mi; }
  for (int i = 0; i < KNN_ * KNN_; ++i) {
    int p = cand[i];
    float dx = xb[p] - cx, dy = xb[N_ + p] - cy, dz = xb[2 * N_ + p] - cz;
    float d = dx * dx + dy * dy + dz * dz;
    if (d < bd[K_]) {
      int j = K_;
      while (j > 0 && d < bd[j - 1]) { bd[j] = bd[j - 1]; bi[j] = bi[j - 1]; --j; }
      bd[j] = d; bi[j] = p;
    }
  }
  int* o = idx + ((size_t)b * S_ + s) * K_;
  #pragma unroll
  for (int k = 0; k < K_; ++k) o[k] = bi[k + 1];
}

// ---------------------------------------------------------------------
// 4) Build GEMM operand X0: [NTOT][96] bf16  (gnorm ++ features ++ zero pad)
// ---------------------------------------------------------------------
__global__ __launch_bounds__(256) void feat_kernel(
    const float* __restrict__ xyz, const float* __restrict__ pts,
    const int* __restrict__ fpsidx, const int* __restrict__ idx,
    __bf16* __restrict__ X0)
{
  int np = blockIdx.x * 256 + threadIdx.x;
  if (np >= NTOT_) return;
  int k = np % K_, bs = np / K_, b = bs / S_;
  int cn = fpsidx[bs];
  int pt = idx[(size_t)bs * K_ + k];
  const float* xb = xyz + (size_t)b * 3 * N_;
  __bf16* o = X0 + (size_t)np * C0PAD_;
  o[0] = (__bf16)(xb[pt] - xb[cn]);
  o[1] = (__bf16)(xb[N_ + pt] - xb[N_ + cn]);
  o[2] = (__bf16)(xb[2 * N_ + pt] - xb[2 * N_ + cn]);
  const float* pb = pts + (size_t)b * D_ * N_;
  #pragma unroll 4
  for (int c = 0; c < D_; ++c) o[3 + c] = (__bf16)pb[c * N_ + pt];
  #pragma unroll
  for (int c = 3 + D_; c < C0PAD_; ++c) o[c] = (__bf16)0.f;
}

// ---------------------------------------------------------------------
// 5) Weight pad + bf16 convert: W[M][C] f32 -> [M][Cpad] bf16
// ---------------------------------------------------------------------
__global__ __launch_bounds__(256) void wpad_kernel(
    const float* __restrict__ w, __bf16* __restrict__ o, int M, int C, int Cpad)
{
  int i = blockIdx.x * 256 + threadIdx.x;
  if (i >= M * Cpad) return;
  int m = i / Cpad, c = i % Cpad;
  o[i] = (__bf16)(c < C ? w[m * C + c] : 0.f);
}

// ---------------------------------------------------------------------
// 6) WMMA bf16 GEMM, full-M strip per wave + fused BN-stats epilogue.
//    Y[n'][m] = sum_c W[m][c] * X[n'][c].
//    One wave owns one 16-column N tile and all MT row tiles:
//      - B operand (X tile) loaded ONCE per K step, reused by MT WMMAs
//      - D tiles reduced over N in-register (shfl butterfly) -> LDS
//        ds_add_f32 partials -> one global atomic per channel per block,
//        removing the separate full-tensor stats read pass.
// ---------------------------------------------------------------------
template <int MT>
__global__ __launch_bounds__(256) void gemm_bf16_fused(
    const __bf16* __restrict__ X,   // [NTOT][Cpad]
    const __bf16* __restrict__ W,   // [MT*16][Cpad]
    float* __restrict__ Y,          // [NTOT][MT*16]
    float* __restrict__ stats,      // [2*MT*16] (sum, then sumsq)
    int Cpad)
{
  constexpr int M = MT * 16;
  __shared__ float ls[2 * M];
  for (int i = threadIdx.x; i < 2 * M; i += 256) ls[i] = 0.f;
  __syncthreads();

  int wv   = threadIdx.x >> 5;                 // 8 wave32 per block
  int nt   = blockIdx.x * 8 + wv;              // 16-column N tile
  int lane = threadIdx.x & 31;
  int half = lane >> 4;                        // high-K / high-M halves
  int lm   = lane & 15;

  const __bf16* xrow  = X + (size_t)(nt * 16 + lm) * Cpad + 16 * half;
  const __bf16* wbase = W + (size_t)lm * Cpad + 16 * half;

  v8f acc[MT];
  #pragma unroll
  for (int t = 0; t < MT; ++t) acc[t] = (v8f){};

  for (int kk = 0; kk < Cpad; kk += 32) {
    // B 32x16 bf16: lane holds column lm, K = 16*half + [0..15] (one 32B load)
    v16bf bmat = *(const v16bf*)(xrow + kk);
    #pragma unroll
    for (int t = 0; t < MT; ++t) {
      // A 16x32 bf16: lane holds row lm of tile t, K = 16*half + [0..15]
      const __bf16* wrow = wbase + (size_t)(t * 16) * Cpad + kk;
      v8bf a0 = *(const v8bf*)(wrow);
      v8bf a1 = *(const v8bf*)(wrow + 8);
      v16bf a = __builtin_shufflevector(a0, a1, 0, 1, 2, 3, 4, 5, 6, 7,
                                                8, 9, 10, 11, 12, 13, 14, 15);
      acc[t] = __builtin_amdgcn_wmma_f32_16x16x32_bf16(
          false, a, false, bmat, (short)0, acc[t], false, false);
    }
  }

  // D f32 16x16: lane lm+16*half holds column N=lm, rows M = t*16 + 8*half + r
  float* yout = Y + (size_t)(nt * 16 + lm) * M + 8 * half;
  #pragma unroll
  for (int t = 0; t < MT; ++t) {
    *(v8f*)(yout + t * 16) = acc[t];
    #pragma unroll
    for (int r = 0; r < 8; ++r) {
      float s  = acc[t][r];
      float s2 = s * s;
      #pragma unroll
      for (int msk = 1; msk < 16; msk <<= 1) {   // reduce over the 16 N columns
        s  += __shfl_xor(s,  msk, 32);
        s2 += __shfl_xor(s2, msk, 32);
      }
      if (lm == 0) {                              // lanes 0 and 16
        atomicAdd(&ls[t * 16 + 8 * half + r], s);
        atomicAdd(&ls[M + t * 16 + 8 * half + r], s2);
      }
    }
  }
  __syncthreads();
  for (int i = threadIdx.x; i < 2 * M; i += 256) atomicAdd(&stats[i], ls[i]);
}

// ---------------------------------------------------------------------
// 7) BN + ReLU -> bf16 activations for the next GEMM  (conv bias is a
//    no-op under train-mode BN and is dropped)
// ---------------------------------------------------------------------
__global__ __launch_bounds__(256) void bnrelu_kernel(
    const float* __restrict__ Y, const float* __restrict__ stats,
    const float* __restrict__ g, const float* __restrict__ be,
    __bf16* __restrict__ Xout, int M, long total)
{
  long i = (long)blockIdx.x * 256 + threadIdx.x;
  if (i >= total) return;
  int m = (int)(i % M);
  long n = i / M;
  const float cnt = (float)NTOT_;
  float mean = stats[m] / cnt;
  float var  = stats[M + m] / cnt - mean * mean;
  float sc   = g[m] * rsqrtf(var + BN_EPS_);
  float v    = (Y[i] - mean) * sc + be[m];
  Xout[n * M + m] = (__bf16)fmaxf(v, 0.f);
}

// ---------------------------------------------------------------------
// 8) Final layer: BN + ReLU + max over K neighbors -> d_out new_points
// ---------------------------------------------------------------------
__global__ __launch_bounds__(256) void bnrelu_maxk_kernel(
    const float* __restrict__ Y, const float* __restrict__ stats,
    const float* __restrict__ g, const float* __restrict__ be,
    float* __restrict__ out)
{
  int t = blockIdx.x * 256 + threadIdx.x;        // (b,s,m), m fastest
  if (t >= B_ * S_ * M2_) return;
  int m = t % M2_, rest = t / M2_, s = rest % S_, b = rest / S_;
  const float cnt = (float)NTOT_;
  float mean = stats[m] / cnt;
  float var  = stats[M2_ + m] / cnt - mean * mean;
  float sc   = g[m] * rsqrtf(var + BN_EPS_);
  float bia  = be[m];
  float mx = 0.f;                                // relu(...) >= 0
  const float* yb = Y + (size_t)((b * S_ + s) * K_) * M2_ + m;
  #pragma unroll 4
  for (int k = 0; k < K_; ++k) {
    float v = (yb[(size_t)k * M2_] - mean) * sc + bia;
    mx = fmaxf(mx, fmaxf(v, 0.f));
  }
  out[(size_t)B_ * 3 * S_ + ((size_t)(b * M2_ + m)) * S_ + s] = mx;
}

// ---------------------------------------------------------------------
extern "C" void kernel_launch(void* const* d_in, const int* in_sizes, int n_in,
                              void* d_out, int out_size, void* d_ws, size_t ws_size,
                              hipStream_t stream) {
  (void)in_sizes; (void)n_in; (void)out_size; (void)ws_size;
  const float* xyz = (const float*)d_in[0];
  const float* pts = (const float*)d_in[1];
  const float* w0  = (const float*)d_in[2];
  const float* g0  = (const float*)d_in[4];
  const float* be0 = (const float*)d_in[5];
  const float* w1  = (const float*)d_in[6];
  const float* g1  = (const float*)d_in[8];
  const float* be1 = (const float*)d_in[9];
  const float* w2  = (const float*)d_in[10];
  const float* g2  = (const float*)d_in[12];
  const float* be2 = (const float*)d_in[13];
  float* out = (float*)d_out;

  // -------- workspace carve-up (≈215 MB) --------
  char* base = (char*)d_ws;
  size_t off = 0;
  auto carve = [&](size_t bytes) { char* p = base + off; off = (off + bytes + 255) & ~(size_t)255; return p; };
  __bf16* XA   = (__bf16*)carve((size_t)NTOT_ * C0PAD_ * 2);   // layer0 in / layer1 out
  __bf16* XB   = (__bf16*)carve((size_t)NTOT_ * C1_ * 2);      // layer0 out / layer1 in
  float*  Y    = (float*) carve((size_t)NTOT_ * M2_ * 4);      // GEMM out, reused
  __bf16* W0p  = (__bf16*)carve((size_t)C1_ * C0PAD_ * 2);
  __bf16* W1p  = (__bf16*)carve((size_t)C1_ * C1_ * 2);
  __bf16* W2p  = (__bf16*)carve((size_t)M2_ * C2_ * 2);
  float*  st   = (float*) carve((size_t)2 * (C1_ + C1_ + M2_) * 4);
  float*  st0 = st, *st1 = st + 2 * C1_, *st2 = st + 4 * C1_;
  int* nbr    = (int*)carve((size_t)B_ * N_ * KNN_ * 4);
  int* maxind = (int*)carve((size_t)B_ * N_ * 4);
  int* fpsidx = (int*)carve((size_t)B_ * S_ * 4);
  int* idx    = (int*)carve((size_t)B_ * S_ * K_ * 4);

  hipMemsetAsync(st, 0, (size_t)2 * (C1_ + C1_ + M2_) * 4, stream);

  // -------- geometry stage --------
  knn_kernel<<<B_ * 8, 256, 0, stream>>>(xyz, nbr, maxind);
  fps_kernel<<<B_, 256, 0, stream>>>(xyz, fpsidx, out);
  expand_kernel<<<(B_ * S_ + 255) / 256, 256, 0, stream>>>(xyz, nbr, maxind, fpsidx, idx);
  feat_kernel<<<(NTOT_ + 255) / 256, 256, 0, stream>>>(xyz, pts, fpsidx, idx, XA);

  // -------- weight conversion --------
  wpad_kernel<<<(C1_ * C0PAD_ + 255) / 256, 256, 0, stream>>>(w0, W0p, C1_, 3 + D_, C0PAD_);
  wpad_kernel<<<(C1_ * C1_ + 255) / 256, 256, 0, stream>>>(w1, W1p, C1_, C1_, C1_);
  wpad_kernel<<<(M2_ * C2_ + 255) / 256, 256, 0, stream>>>(w2, W2p, M2_, C2_, C2_);

  const int NT16   = NTOT_ / 16;     // 16384 column tiles
  const int GBLKS  = NT16 / 8;       // 2048 blocks, 8 waves (tiles) each

  // -------- layer 0: 96 -> 64 --------
  gemm_bf16_fused<4><<<GBLKS, 256, 0, stream>>>(XA, W0p, Y, st0, C0PAD_);
  bnrelu_kernel<<<(int)(((long)NTOT_ * C1_ + 255) / 256), 256, 0, stream>>>(
      Y, st0, g0, be0, XB, C1_, (long)NTOT_ * C1_);

  // -------- layer 1: 64 -> 64 --------
  gemm_bf16_fused<4><<<GBLKS, 256, 0, stream>>>(XB, W1p, Y, st1, C1_);
  bnrelu_kernel<<<(int)(((long)NTOT_ * C1_ + 255) / 256), 256, 0, stream>>>(
      Y, st1, g1, be1, XA, C1_, (long)NTOT_ * C1_);   // XA reused, row stride 64

  // -------- layer 2: 64 -> 128, then max over K --------
  gemm_bf16_fused<8><<<GBLKS, 256, 0, stream>>>(XA, W2p, Y, st2, C2_);
  bnrelu_maxk_kernel<<<(B_ * S_ * M2_ + 255) / 256, 256, 0, stream>>>(Y, st2, g2, be2, out);
}
